// CenterTOp2_5617817223883
// MI455X (gfx1250) — compile-verified, matrix operand order checked by reading, not run on hardware
//
#include <hip/hip_runtime.h>

typedef __attribute__((ext_vector_type(16))) _Float16 v16h;
typedef __attribute__((ext_vector_type(8)))  float    v8f;

#define KCLS 2
#define CCH  256
#define NPIX 65536
#define BB   8
#define ROWS_PER_BLOCK 128
#define THRESH_C 0.01f

// workspace float offsets
#define WS_CENTERS 0      // 2*256
#define WS_CNORM   512    // 2*256 (normalized centers used by assign)
#define WS_SUMS    1024   // 8*2*256
#define WS_COUNTS  5120   // 8*2
#define WS_STATE   5136   // [cur_dist, done]

// d_out float offsets (return order: centers, labels, onehot, cur_dist)
#define OUT_CENTERS 0
#define OUT_LABELS  512
#define OUT_ONEHOT  (512 + BB * NPIX)             // 524800
#define OUT_CURDIST (OUT_ONEHOT + BB * NPIX * 2)  // 1573376

__device__ __forceinline__ float block_reduce_256(float v, float* sb) {
    int t = threadIdx.x;
    sb[t] = v;
    __syncthreads();
#pragma unroll
    for (int s = 128; s > 0; s >>= 1) {
        if (t < s) sb[t] += sb[t + s];
        __syncthreads();
    }
    float r = sb[0];
    __syncthreads();
    return r;
}

__global__ void ctop_init(const float* __restrict__ centerInit, float* __restrict__ ws) {
    __shared__ float sb[256];
    int c = threadIdx.x;
    float c0 = centerInit[c];
    float c1 = centerInit[CCH + c];
    ws[WS_CENTERS + c]       = c0;
    ws[WS_CENTERS + CCH + c] = c1;
    float n0 = sqrtf(block_reduce_256(c0 * c0, sb));
    float n1 = sqrtf(block_reduce_256(c1 * c1, sb));
    ws[WS_CNORM + c]       = c0 / fmaxf(n0, 1e-12f);
    ws[WS_CNORM + CCH + c] = c1 / fmaxf(n1, 1e-12f);
    if (c == 0) {
        ws[WS_STATE + 0] = 1e9f;  // cur_dist (overwritten at step 1)
        ws[WS_STATE + 1] = 0.0f;  // done flag
    }
}

__global__ void ctop_zero(float* __restrict__ ws) {
    int i = blockIdx.x * 256 + threadIdx.x;
    if (i < BB * KCLS * CCH + BB * KCLS) ws[WS_SUMS + i] = 0.0f;
}

// One block = 128 rows of one batch. 8 waves; each wave WMMA-labels 16 rows,
// then all 256 threads accumulate per-channel partial sums (thread t owns
// channel t exclusively) and flush with f32 global atomics.
__global__ void __launch_bounds__(256) ctop_assign(const float* __restrict__ feats,
                                                   float* __restrict__ ws,
                                                   float* __restrict__ out) {
    __shared__ _Float16 s_b[16 * CCH];   // zero-padded B panel: [col 0..15][K 0..255]
    __shared__ int s_lbl[ROWS_PER_BLOCK];

    const int b  = blockIdx.y;
    const int n0 = blockIdx.x * ROWS_PER_BLOCK;
    const int tid  = threadIdx.x;
    const int lane = tid & 31;
    const int w    = tid >> 5;
    const int half = lane >> 4;   // which K-group this lane holds
    const int m    = lane & 15;   // A row (M) / B column (N) for this lane
    const int kbase = half * 8;

    const float* __restrict__ Xb    = feats + (size_t)b * CCH * NPIX;  // [C][N]
    const float* __restrict__ cnorm = ws + WS_CNORM;
    const bool done = (ws[WS_STATE + 1] != 0.0f);

    // Stage zero-padded B panel into LDS (cols 0/1 = normalized centers, rest 0).
#pragma unroll
    for (int i = 0; i < 16; ++i) {
        int idx = tid + 256 * i;          // 0..4095
        int col = idx >> 8;
        int kk  = idx & 255;
        float v = (col < KCLS) ? cnorm[col * CCH + kk] : 0.0f;
        s_b[idx] = (_Float16)v;
    }
    __syncthreads();

    // ---- Phase 1: dots = feats_tile[16 x 256] * B  via 8x v_wmma_f32_16x16x32_f16
    const int nrow = n0 + w * 16 + m;     // this lane's A row in the image
    const _Float16* __restrict__ bcol = s_b + m * CCH;
    v8f acc = {};
#pragma unroll
    for (int q = 0; q < 8; ++q) {
        const int c0 = q * 32;
        const float* __restrict__ ap = Xb + (size_t)c0 * NPIX + nrow;
        v16h av, bv;
#pragma unroll
        for (int j = 0; j < 16; ++j) {
            const int Kj = kbase + (j < 8 ? j : j + 8);  // K in {kbase..+7, 16+kbase..}
            av[j] = (_Float16)ap[(size_t)Kj * NPIX];     // const ioffset (< 8 MB)
            bv[j] = bcol[c0 + Kj];                       // branchless ds_load
        }
        acc = __builtin_amdgcn_wmma_f32_16x16x32_f16(false, av, false, bv,
                                                     (short)0, acc, false, false);
    }

    // D layout: VGPR v -> lanes 0-15: (M=v, N=lane); lanes 16-31: (M=8+v, N=lane-16)
#pragma unroll
    for (int v = 0; v < 8; ++v) {
        float mine  = acc[v];                 // dot with center N=(lane&15)
        float other = __shfl_xor(mine, 1, 32);
        if (m == 0) {                         // lanes 0 (rows v) and 16 (rows 8+v)
            int row = w * 16 + half * 8 + v;
            s_lbl[row] = (other > mine) ? 1 : 0;  // argmin dist == argmax dot; tie -> 0
        }
    }
    __syncthreads();

    // labels output (frozen once converged, matching the reference while-loop)
    if (tid < ROWS_PER_BLOCK && !done)
        out[OUT_LABELS + (size_t)b * NPIX + n0 + tid] = (float)s_lbl[tid];

    // ---- Phase 2: per-channel masked sums; tile was just fetched -> L2 hit
    const int c = tid;  // 0..255, exclusive channel owner
    float a0 = 0.0f, a1 = 0.0f;
    const float* rowp = Xb + (size_t)c * NPIX + n0;  // 128 contiguous, 16B aligned
#pragma unroll 4
    for (int r = 0; r < ROWS_PER_BLOCK; r += 4) {
        float4 v = *(const float4*)(rowp + r);
        float l0 = (float)s_lbl[r + 0], l1 = (float)s_lbl[r + 1];
        float l2 = (float)s_lbl[r + 2], l3 = (float)s_lbl[r + 3];
        a1 += l0 * v.x + l1 * v.y + l2 * v.z + l3 * v.w;
        a0 += (1.0f - l0) * v.x + (1.0f - l1) * v.y + (1.0f - l2) * v.z + (1.0f - l3) * v.w;
    }
    atomicAdd(&ws[WS_SUMS + (b * KCLS + 0) * CCH + c], a0);
    atomicAdd(&ws[WS_SUMS + (b * KCLS + 1) * CCH + c], a1);
    if (tid == 0) {
        int c1 = 0;
        for (int r = 0; r < ROWS_PER_BLOCK; ++r) c1 += s_lbl[r];
        atomicAdd(&ws[WS_COUNTS + b * KCLS + 1], (float)c1);
        atomicAdd(&ws[WS_COUNTS + b * KCLS + 0], (float)(ROWS_PER_BLOCK - c1));
    }
}

__global__ void ctop_finalize(float* __restrict__ ws) {
    __shared__ float sb[256];
    const int c = threadIdx.x;
    const float done_old = ws[WS_STATE + 1];

    float cOld0 = ws[WS_CENTERS + c];
    float cOld1 = ws[WS_CENTERS + CCH + c];
    float cn0 = sqrtf(block_reduce_256(cOld0 * cOld0, sb));
    float cn1 = sqrtf(block_reduce_256(cOld1 * cOld1, sb));

    float newc0 = 0.0f, newc1 = 0.0f, cdist_sum = 0.0f;
    for (int b = 0; b < BB; ++b) {
        float cd = 0.0f;
#pragma unroll
        for (int k = 0; k < KCLS; ++k) {
            float cnt = ws[WS_COUNTS + b * KCLS + k];
            float ci  = ws[WS_SUMS + (b * KCLS + k) * CCH + c] / (cnt + 1.0f);
            float cok = (k == 0) ? cOld0 : cOld1;
            float num = block_reduce_256(ci * cok, sb);
            float nrm = sqrtf(block_reduce_256(ci * ci, sb));
            float den = fmaxf(nrm * ((k == 0) ? cn0 : cn1), 1e-8f);
            cd += num / den;
            if (k == 0) newc0 += ci; else newc1 += ci;
        }
        cdist_sum += cd * (1.0f / (float)KCLS);
    }
    float cur = cdist_sum / (float)BB;

    if (done_old == 0.0f) {
        ws[WS_CENTERS + c]       = newc0 / (float)BB;
        ws[WS_CENTERS + CCH + c] = newc1 / (float)BB;
    }
    __syncthreads();

    float u0 = ws[WS_CENTERS + c];
    float u1 = ws[WS_CENTERS + CCH + c];
    float m0 = sqrtf(block_reduce_256(u0 * u0, sb));
    float m1 = sqrtf(block_reduce_256(u1 * u1, sb));
    ws[WS_CNORM + c]       = u0 / fmaxf(m0, 1e-12f);
    ws[WS_CNORM + CCH + c] = u1 / fmaxf(m1, 1e-12f);

    if (c == 0) {
        if (done_old == 0.0f) ws[WS_STATE + 0] = cur;
        ws[WS_STATE + 1] = (done_old != 0.0f || ws[WS_STATE + 0] < THRESH_C) ? 1.0f : 0.0f;
    }
}

__global__ void ctop_emit(const float* __restrict__ ws, float* __restrict__ out) {
    size_t i = (size_t)blockIdx.x * 256 + threadIdx.x;  // 0 .. B*N-1
    float l = out[OUT_LABELS + i];
    out[OUT_ONEHOT + 2 * i + 0] = 1.0f - l;
    out[OUT_ONEHOT + 2 * i + 1] = l;
    if (i < KCLS * CCH) out[OUT_CENTERS + i] = ws[WS_CENTERS + i];
    if (i == 0) out[OUT_CURDIST] = ws[WS_STATE + 0];
}

extern "C" void kernel_launch(void* const* d_in, const int* in_sizes, int n_in,
                              void* d_out, int out_size, void* d_ws, size_t ws_size,
                              hipStream_t stream) {
    (void)in_sizes; (void)n_in; (void)out_size; (void)ws_size;
    const float* feats      = (const float*)d_in[0];   // [8,256,256,256] = [B][C][N]
    const float* centerInit = (const float*)d_in[1];   // [2,256]
    float* ws  = (float*)d_ws;
    float* out = (float*)d_out;

    ctop_init<<<1, 256, 0, stream>>>(centerInit, ws);
    for (int it = 0; it < 3; ++it) {
        ctop_zero<<<17, 256, 0, stream>>>(ws);
        ctop_assign<<<dim3(NPIX / ROWS_PER_BLOCK, BB), 256, 0, stream>>>(feats, ws, out);
        ctop_finalize<<<1, 256, 0, stream>>>(ws);
    }
    ctop_emit<<<(BB * NPIX) / 256, 256, 0, stream>>>(ws, out);
}
